// GATEncoder_29557964931453
// MI455X (gfx1250) — matrix operand order repaired and checked by reading, not compiled, not run on hardware
//
#include <hip/hip_runtime.h>
#include <hip/hip_fp16.h>
#include <cstdint>
#include <cstddef>

// ---------------- WMMA types (probe-verified for gfx1250) ----------------
typedef __attribute__((ext_vector_type(16))) _Float16 v16h;
typedef __attribute__((ext_vector_type(8)))  float    v8f;

#define TPB 256

// ---------------- small elementwise helpers ----------------
__global__ void k_f32_to_f16(const float* __restrict__ in, _Float16* __restrict__ out, int n) {
  int i = blockIdx.x * blockDim.x + threadIdx.x;
  if (i < n) out[i] = (_Float16)in[i];
}

// W: K x M row-major f32  ->  Wt: M x K row-major f16 (so B fragments load contiguously)
__global__ void k_w_transpose_f16(const float* __restrict__ W, _Float16* __restrict__ Wt,
                                  int K, int M) {
  int i = blockIdx.x * blockDim.x + threadIdx.x;
  if (i < K * M) {
    int k = i / M, m = i - k * M;
    Wt[(size_t)m * K + k] = (_Float16)W[i];
  }
}

__global__ void k_fill(float* __restrict__ p, float v, int n) {
  int i = blockIdx.x * blockDim.x + threadIdx.x;
  if (i < n) p[i] = v;
}

// ---------------- WMMA GEMM: Y[N x M] = f16(X)[N x K] @ W[K x M] + bias ----------------
// Wt is the M x K (row-major) f16 transpose of W. One wave computes one 16x16 tile of Y.
// N multiple of 16, K multiple of 32, M multiple of 16 (all exact for this model).
__global__ void k_wmma_gemm_bias(const _Float16* __restrict__ Xh,
                                 const _Float16* __restrict__ Wt,
                                 const float* __restrict__ bias,
                                 float* __restrict__ Y,
                                 int Nrows, int K, int M) {
  const int wave = (int)((blockIdx.x * blockDim.x + threadIdx.x) >> 5);
  const int lane = threadIdx.x & 31;
  const int tilesM = M >> 4;
  const int nTiles = (Nrows >> 4) * tilesM;
  if (wave >= nTiles) return;          // whole-wave uniform: EXEC stays all-1 for WMMA
  const int tr = wave / tilesM;
  const int tc = wave - tr * tilesM;

  const int hi  = lane >> 4;           // 0: lanes 0-15, 1: lanes 16-31
  const int l15 = lane & 15;

  const int arow = tr * 16 + l15;      // A-matrix row handled by this lane
  const int bcol = tc * 16 + l15;      // B-matrix column handled by this lane

  v8f acc = {};
  for (int k0 = 0; k0 < K; k0 += 32) {
    union { v16h v; uint32_t u[8]; } A, B;
    // A 16x32 f16 layout: lanes 0-15 -> K 0..15; lanes 16-31 -> K 16..31
    const uint32_t* a32 =
        reinterpret_cast<const uint32_t*>(Xh + (size_t)arow * K + k0 + hi * 16);
    // B 32x16 f16 layout: lanes 0-15 -> K 0..15; lanes 16-31 -> K 16..31 (contiguous in Wt)
    const uint32_t* b32 =
        reinterpret_cast<const uint32_t*>(Wt + (size_t)bcol * K + k0 + hi * 16);
#pragma unroll
    for (int v = 0; v < 4; ++v) { A.u[v] = a32[v]; A.u[v + 4] = a32[v + 4]; }
#pragma unroll
    for (int v = 0; v < 8; ++v) { B.u[v] = b32[v]; }
    acc = __builtin_amdgcn_wmma_f32_16x16x32_f16(
        /*neg_a=*/false, A.v, /*neg_b=*/false, B.v,
        /*c_mod=*/(short)0, acc, /*reuse_a=*/false, /*reuse_b=*/false);
  }
  // D layout: lane l15 = column, VGPR v -> row v + 8*hi
  const float bv = bias[tc * 16 + l15];
#pragma unroll
  for (int v = 0; v < 8; ++v) {
    int m = v + hi * 8;
    Y[(size_t)(tr * 16 + m) * M + tc * 16 + l15] = acc[v] + bv;
  }
}

// ---------------- edge-phase kernels ----------------
__device__ __forceinline__ int e_src(const int* __restrict__ ei, int e, int E) {
  return (e < E) ? ei[e] : (e - E);            // self-loops appended implicitly
}
__device__ __forceinline__ int e_dst(const int* __restrict__ ei, int e, int E) {
  return (e < E) ? ei[E + e] : (e - E);
}

__device__ __forceinline__ void atomicMaxF(float* addr, float val) {
  // sign-aware int/uint trick: correct total order on float bit patterns
  if (val >= 0.0f) atomicMax((int*)addr, __float_as_int(val));
  else             atomicMin((unsigned int*)addr, __float_as_uint(val));
}

// logit[e,h] = sum_c att[h,c] * leaky_relu(xl[src,h,c] + xr[dst,h,c]); seg-max into mbuf
__global__ void k_edge_logit(const int* __restrict__ ei, int E, int EP,
                             const float* __restrict__ xl, const float* __restrict__ xr,
                             const float* __restrict__ att,
                             float* __restrict__ logit, float* __restrict__ mbuf,
                             int H, int C, int M) {
  int t = blockIdx.x * blockDim.x + threadIdx.x;
  if (t >= EP * H) return;
  int e = t / H, h = t - e * H;
  int s = e_src(ei, e, E), d = e_dst(ei, e, E);
  const float* xa = xl + (size_t)s * M + h * C;
  const float* xb = xr + (size_t)d * M + h * C;
  const float* at = att + h * C;
  float acc = 0.0f;
  for (int c = 0; c < C; ++c) {
    float v = xa[c] + xb[c];
    v = (v > 0.0f) ? v : 0.2f * v;             // leaky_relu, slope 0.2
    acc += v * at[c];
  }
  logit[t] = acc;
  atomicMaxF(&mbuf[(size_t)d * H + h], acc);
}

// exb[t] = exp(logit - m[dst]); seg-sum into denom   (logit buffer updated in place)
__global__ void k_edge_exp(const int* __restrict__ ei, int E, int EP,
                           const float* __restrict__ mbuf,
                           float* __restrict__ exb, float* __restrict__ denom, int H) {
  int t = blockIdx.x * blockDim.x + threadIdx.x;
  if (t >= EP * H) return;
  int e = t / H, h = t - e * H;
  int d = e_dst(ei, e, E);
  float ex = __expf(exb[t] - mbuf[(size_t)d * H + h]);
  exb[t] = ex;
  atomicAdd(&denom[(size_t)d * H + h], ex);
}

// agg[dst,h,:] += alpha * xl[src,h,:]
__global__ void k_edge_agg(const int* __restrict__ ei, int E, int EP,
                           const float* __restrict__ xl, const float* __restrict__ exb,
                           const float* __restrict__ denom, float* __restrict__ agg,
                           int H, int C, int M) {
  int t = blockIdx.x * blockDim.x + threadIdx.x;
  if (t >= EP * H) return;
  int e = t / H, h = t - e * H;
  int s = e_src(ei, e, E), d = e_dst(ei, e, E);
  float a = exb[t] / (denom[(size_t)d * H + h] + 1e-16f);
  const float* xa = xl + (size_t)s * M + h * C;
  float* o = agg + (size_t)d * M + h * C;
  for (int c = 0; c < C; ++c) atomicAdd(&o[c], xa[c] * a);
}

// out = [prev +] relu(agg + bias)   (concat layers)
__global__ void k_finalize_concat(const float* __restrict__ agg, const float* __restrict__ bias,
                                  const float* __restrict__ prev, int addPrev,
                                  float* __restrict__ out, int n, int M) {
  int i = blockIdx.x * blockDim.x + threadIdx.x;
  if (i >= n) return;
  float v = agg[i] + bias[i % M];
  v = (v > 0.0f) ? v : 0.0f;
  out[i] = addPrev ? prev[i] + v : v;
}

// out[n,c] = relu(mean_h agg[n,h,c] + bias[c])   (final layer, concat=False)
__global__ void k_finalize_mean(const float* __restrict__ agg, const float* __restrict__ bias,
                                float* __restrict__ out, int N, int H, int C) {
  int i = blockIdx.x * blockDim.x + threadIdx.x;
  if (i >= N * C) return;
  int n = i / C, c = i - n * C;
  float s = 0.0f;
  for (int h = 0; h < H; ++h) s += agg[(size_t)n * H * C + h * C + c];
  float v = s * (1.0f / H) + bias[c];
  out[i] = (v > 0.0f) ? v : 0.0f;
}

// ---------------- host orchestration ----------------
static inline int blks(long long n) { return (int)((n + TPB - 1) / TPB); }

namespace {
constexpr int N_  = 50000;
constexpr int E_  = 800000;
constexpr int EP_ = E_ + N_;   // with self-loops
constexpr int H_  = 4;
}

struct Scratch {
  float *xl, *xr, *exb, *mbuf, *denom, *agg, *h0, *h1;
  _Float16 *xh, *wlt, *wrt;
};

static void gat_layer(const float* X, int K, int C, int M,
                      const float* Wl, const float* Wr,
                      const float* bl, const float* br, const float* att,
                      const int* ei, const Scratch& s, hipStream_t st) {
  // feature + weight precision conversion
  k_f32_to_f16<<<blks((long long)N_ * K), TPB, 0, st>>>(X, s.xh, N_ * K);
  k_w_transpose_f16<<<blks((long long)K * M), TPB, 0, st>>>(Wl, s.wlt, K, M);
  k_w_transpose_f16<<<blks((long long)K * M), TPB, 0, st>>>(Wr, s.wrt, K, M);
  // xl = f16(X) @ Wl + bl ; xr = f16(X) @ Wr + br   (WMMA)
  int nTiles = (N_ / 16) * (M / 16);
  int gB = (nTiles + (TPB / 32) - 1) / (TPB / 32);
  k_wmma_gemm_bias<<<gB, TPB, 0, st>>>(s.xh, s.wlt, bl, s.xl, N_, K, M);
  k_wmma_gemm_bias<<<gB, TPB, 0, st>>>(s.xh, s.wrt, br, s.xr, N_, K, M);
  // segment-softmax state
  k_fill<<<blks((long long)N_ * H_), TPB, 0, st>>>(s.mbuf, -3.0e38f, N_ * H_);
  k_fill<<<blks((long long)N_ * H_), TPB, 0, st>>>(s.denom, 0.0f, N_ * H_);
  k_fill<<<blks((long long)N_ * M), TPB, 0, st>>>(s.agg, 0.0f, N_ * M);
  // edge phase
  long long nt = (long long)EP_ * H_;
  k_edge_logit<<<blks(nt), TPB, 0, st>>>(ei, E_, EP_, s.xl, s.xr, att, s.exb, s.mbuf, H_, C, M);
  k_edge_exp  <<<blks(nt), TPB, 0, st>>>(ei, E_, EP_, s.mbuf, s.exb, s.denom, H_);
  k_edge_agg  <<<blks(nt), TPB, 0, st>>>(ei, E_, EP_, s.xl, s.exb, s.denom, s.agg, H_, C, M);
}

extern "C" void kernel_launch(void* const* d_in, const int* in_sizes, int n_in,
                              void* d_out, int out_size, void* d_ws, size_t ws_size,
                              hipStream_t stream) {
  (void)in_sizes; (void)n_in; (void)out_size; (void)ws_size;
  const float* x    = (const float*)d_in[0];
  const int*   ei   = (const int*)  d_in[1];   // edge_index [2,E]
  const float* Wl0  = (const float*)d_in[2];
  const float* Wr0  = (const float*)d_in[3];
  const float* bl0  = (const float*)d_in[4];
  const float* br0  = (const float*)d_in[5];
  const float* att0 = (const float*)d_in[6];
  const float* bi0  = (const float*)d_in[7];
  const float* Wl1  = (const float*)d_in[8];
  const float* Wr1  = (const float*)d_in[9];
  const float* bl1  = (const float*)d_in[10];
  const float* br1  = (const float*)d_in[11];
  const float* att1 = (const float*)d_in[12];
  const float* bi1  = (const float*)d_in[13];
  const float* WlL  = (const float*)d_in[14];
  const float* WrL  = (const float*)d_in[15];
  const float* blL  = (const float*)d_in[16];
  const float* brL  = (const float*)d_in[17];
  const float* attL = (const float*)d_in[18];
  const float* biL  = (const float*)d_in[19];
  float* out = (float*)d_out;

  // workspace carve-up (256B-aligned slabs)
  char* base = (char*)d_ws;
  size_t off = 0;
  auto take = [&](size_t bytes) -> void* {
    void* p = base + off;
    off += (bytes + 255) & ~(size_t)255;
    return p;
  };
  Scratch s;
  s.xl    = (float*)take((size_t)N_ * 256 * 4);
  s.xr    = (float*)take((size_t)N_ * 256 * 4);
  s.exb   = (float*)take((size_t)EP_ * H_ * 4);
  s.mbuf  = (float*)take((size_t)N_ * H_ * 4);
  s.denom = (float*)take((size_t)N_ * H_ * 4);
  s.agg   = (float*)take((size_t)N_ * 256 * 4);
  s.h0    = (float*)take((size_t)N_ * 128 * 4);
  s.h1    = (float*)take((size_t)N_ * 128 * 4);
  s.xh    = (_Float16*)take((size_t)N_ * 128 * 2);
  s.wlt   = (_Float16*)take((size_t)256 * 128 * 2);
  s.wrt   = (_Float16*)take((size_t)256 * 128 * 2);

  // layer 0: 64 -> 4x32 concat, ReLU
  gat_layer(x, 64, 32, 128, Wl0, Wr0, bl0, br0, att0, ei, s, stream);
  k_finalize_concat<<<blks((long long)N_ * 128), TPB, 0, stream>>>(
      s.agg, bi0, s.agg, 0, s.h0, N_ * 128, 128);

  // layer 1 (residual): 128 -> 4x32 concat, h1 = h0 + relu(layer(h0))
  gat_layer(s.h0, 128, 32, 128, Wl1, Wr1, bl1, br1, att1, ei, s, stream);
  k_finalize_concat<<<blks((long long)N_ * 128), TPB, 0, stream>>>(
      s.agg, bi1, s.h0, 1, s.h1, N_ * 128, 128);

  // final layer: 128 -> 4x64, mean over heads, +bias, ReLU
  gat_layer(s.h1, 128, 64, 256, WlL, WrL, blL, brL, attL, ei, s, stream);
  k_finalize_mean<<<blks((long long)N_ * 64), TPB, 0, stream>>>(
      s.agg, biL, out, N_, H_, 64);
}
